// Attention_84310208020626
// MI455X (gfx1250) — matrix operand order repaired and checked by reading, not compiled, not run on hardware
//
#include <hip/hip_runtime.h>
#include <hip/hip_bf16.h>
#include <stdint.h>

// Problem constants (fixed by the reference)
#define S_LEN 2048
#define BATCH 4
#define DIM   512
#define LN_EPS 1e-5f
#define SOFTMAX_SCALE 0.04419417382415922f   // 1/sqrt(512)

typedef uint16_t u16;
typedef __attribute__((ext_vector_type(16))) __bf16 v16bf;
typedef __attribute__((ext_vector_type(8)))  float  v8f;
typedef __attribute__((ext_vector_type(4)))  unsigned int u32x4;
typedef __attribute__((ext_vector_type(8)))  int          i32x8;
typedef __attribute__((ext_vector_type(4)))  int          i32x4;

#if __has_builtin(__builtin_amdgcn_tensor_load_to_lds) && __has_builtin(__builtin_amdgcn_s_wait_tensorcnt)
#define USE_TDM 1
#else
#define USE_TDM 0
#endif

// ---------------------------------------------------------------- helpers
__device__ __forceinline__ u16 f32_to_bf16_rne(float f) {
  union { float f; uint32_t u; } c; c.f = f;
  uint32_t u = c.u;
  u += 0x7FFFu + ((u >> 16) & 1u);
  return (u16)(u >> 16);
}

union FragU { v16bf v; uint4 q[2]; };

// A fragment: 16x32 bf16 tile of row-major matrix, origin (m0,k0).
// ISA layout: lanes 0-15 -> M=lane, K={k0..k0+7, k0+16..k0+23}; lanes 16-31 -> K={+8, +24}
__device__ __forceinline__ v16bf load_frag_a(const u16* __restrict__ A, int lda,
                                             int m0, int k0, int lane) {
  const int row = m0 + (lane & 15);
  const int kh  = (lane >> 4) << 3;           // 0 or 8
  const u16* p = A + (long)row * lda + k0 + kh;
  FragU f;
  f.q[0] = *(const uint4*)(p);                // K offsets kh .. kh+7
  f.q[1] = *(const uint4*)(p + 16);           // K offsets kh+16 .. kh+23
  return f.v;
}

// ---------------------------------------------------------------- TDM tile feeder
// Stage a (64 rows x 32 K) bf16 tile of the (N,K) row-major B matrix into LDS.
#if USE_TDM
__device__ __forceinline__ void tdm_issue_tile(const u16* gsrc, int ld_elems, u16* lds_dst) {
  const uint64_t ga = (uint64_t)(uintptr_t)gsrc;
  const uint32_t la = (uint32_t)(uintptr_t)lds_dst;   // LDS byte offset = low 32 bits

  u32x4 g0;
  g0.x = 1u;                                           // count=1 (valid), user mode, no gather
  g0.y = la;                                           // lds_addr
  g0.z = (uint32_t)ga;                                 // global_addr[31:0]
  g0.w = (uint32_t)((ga >> 32) & 0x01FFFFFFu) | (2u << 30);  // global_addr[56:32] | type=2

  const int cols = 32, rows = 64;                      // tile/tensor dims (no OOB)
  i32x8 g1;
  g1[0] = 0x00010000;                                  // wg_mask=0, data_size=1 (2 bytes)
  g1[1] = (int)((cols & 0xFFFF) << 16);                // tensor_dim0[15:0]
  g1[2] = (int)(((cols >> 16) & 0xFFFF) | ((rows & 0xFFFF) << 16)); // dim0 hi | dim1 lo
  g1[3] = (int)(((rows >> 16) & 0xFFFF) | ((cols & 0xFFFF) << 16)); // dim1 hi | tile_dim0
  g1[4] = (int)(rows & 0xFFFF);                        // tile_dim1 | tile_dim2=0
  g1[5] = ld_elems;                                    // tensor_dim0_stride[31:0]
  g1[6] = 0;                                           // stride hi | dim1_stride lo
  g1[7] = 0;
  i32x4 z = {};
#if __clang_major__ >= 23
  i32x8 z8 = {};
  __builtin_amdgcn_tensor_load_to_lds(g0, g1, z, z, z8, 0);
#else
  __builtin_amdgcn_tensor_load_to_lds(g0, g1, z, z, 0);
#endif
}
#endif

__device__ __forceinline__ void issue_tile(u16* lds_dst, const u16* gsrc, int ld_elems,
                                           int wave) {
#if USE_TDM
  if (wave == 0) tdm_issue_tile(gsrc, ld_elems, lds_dst);
#else
  // cooperative copy: 256 threads x 16B = 4KB tile (64x32 u16)
  const int t = threadIdx.x;
  const int r = t >> 2, c = (t & 3) * 8;
  *(uint4*)&lds_dst[r * 32 + c] = *(const uint4*)&gsrc[(long)r * ld_elems + c];
  (void)wave;
#endif
}

__device__ __forceinline__ void wait_tile(int wave, bool more_in_flight) {
#if USE_TDM
  if (wave == 0) {
    if (more_in_flight) __builtin_amdgcn_s_wait_tensorcnt((short)1);
    else                __builtin_amdgcn_s_wait_tensorcnt((short)0);
  }
#else
  (void)wave; (void)more_in_flight;
#endif
}

// B fragment from the LDS-resident 64x32 tile.
// ISA layout: lane -> N=lane&15 (+nlocal); lanes 0-15 K={0..7,16..23}, lanes 16-31 K={8..15,24..31}.
__device__ __forceinline__ v16bf load_frag_b_lds(const u16* tile, int nlocal, int lane) {
  const int col = nlocal + (lane & 15);
  const int kh  = (lane >> 4) << 3;           // 0 or 8
  const u16* p = tile + col * 32 + kh;        // K offsets kh..kh+7, kh+16..kh+23
  FragU f;
  f.q[0] = *(const uint4*)(p);
  f.q[1] = *(const uint4*)(p + 16);
  return f.v;
}

// ---------------------------------------------------------------- conversions
__global__ __launch_bounds__(256) void cvt_x_kernel(const float* __restrict__ x,
                                                    u16* __restrict__ xbf) {
  // x: (S,B,D) f32 -> xbf: (B,S,D) bf16
  long i = (long)blockIdx.x * 256 + threadIdx.x;
  int d = (int)(i & (DIM - 1));
  int b = (int)((i >> 9) & (BATCH - 1));
  long s = i >> 11;
  xbf[((long)b * S_LEN + s) * DIM + d] = f32_to_bf16_rne(x[i]);
}

__global__ __launch_bounds__(256) void cvt_w_kernel(const float* __restrict__ src,
                                                    u16* __restrict__ dst, int n) {
  int i = blockIdx.x * 256 + threadIdx.x;
  if (i < n) dst[i] = f32_to_bf16_rne(src[i]);
}

// ---------------------------------------------------------------- WMMA GEMM
// C(MxN) = epilogue( A_bf16(MxK, lda) @ B_bf16(NxK, ldb)^T )
// Block = 8 waves; macro tile 128(M) x 64(N); wave strip 16x64.
// B strip staged in LDS by the Tensor Data Mover, double-buffered 64x32 tiles.
// MODE 0: bf16 out        MODE 1: f32 out        MODE 2: bf16 out, vT layout
// MODE 3: f32 out +bias, final (S,B,D) transpose  MODE 4: bf16 out +bias +ReLU
template <int MODE>
__global__ __launch_bounds__(256) void gemm_bf16_kernel(
    const u16* __restrict__ A, long sA,
    const u16* __restrict__ Bm, long sB,
    void* __restrict__ Cout, long sC,
    int K, int lda, int ldb, int ldc,
    const float* __restrict__ bias) {
  __shared__ __align__(16) u16 bsm[2][64 * 32];   // 2 x 4KB tiles

  const int lane = threadIdx.x & 31;
  const int wave = threadIdx.x >> 5;
  const int m0 = blockIdx.y * 128 + wave * 16;
  const int n0 = blockIdx.x * 64;
  const u16* Ab = A  + (long)blockIdx.z * sA;
  const u16* Bstrip = Bm + (long)blockIdx.z * sB + (long)n0 * ldb;

  // pipeline prologue: tiles for k0=0 and k0=32 in flight
  issue_tile(&bsm[0][0], Bstrip,      ldb, wave);
  issue_tile(&bsm[1][0], Bstrip + 32, ldb, wave);

  v16bf a_cur = load_frag_a(Ab, lda, m0, 0, lane);
  v8f acc[4] = {};

  for (int k0 = 0; k0 < K; k0 += 32) {
    const int cur = (k0 >> 5) & 1;
    wait_tile(wave, k0 + 32 < K);   // oldest in-flight tile (= cur) complete
    __syncthreads();                // bsm[cur] visible to all waves

    v16bf a_nxt = a_cur;
    if (k0 + 32 < K) a_nxt = load_frag_a(Ab, lda, m0, k0 + 32, lane);  // overlap

    v16bf b0 = load_frag_b_lds(&bsm[cur][0], 0,  lane);
    v16bf b1 = load_frag_b_lds(&bsm[cur][0], 16, lane);
    v16bf b2 = load_frag_b_lds(&bsm[cur][0], 32, lane);
    v16bf b3 = load_frag_b_lds(&bsm[cur][0], 48, lane);
    acc[0] = __builtin_amdgcn_wmma_f32_16x16x32_bf16(false, a_cur, false, b0, (short)0, acc[0], false, false);
    acc[1] = __builtin_amdgcn_wmma_f32_16x16x32_bf16(false, a_cur, false, b1, (short)0, acc[1], false, false);
    acc[2] = __builtin_amdgcn_wmma_f32_16x16x32_bf16(false, a_cur, false, b2, (short)0, acc[2], false, false);
    acc[3] = __builtin_amdgcn_wmma_f32_16x16x32_bf16(false, a_cur, false, b3, (short)0, acc[3], false, false);

    __syncthreads();                // all waves done reading bsm[cur]
    if (k0 + 64 < K) issue_tile(&bsm[cur][0], Bstrip + (k0 + 64), ldb, wave);
    a_cur = a_nxt;
  }

  // C/D layout: lane -> N=lane&15; VGPR j -> M = j + 8*(lane>=16)
  const int mrow = m0 + ((lane >> 4) << 3);
  const int ncol = n0 + (lane & 15);
#pragma unroll
  for (int t = 0; t < 4; ++t) {
#pragma unroll
    for (int j = 0; j < 8; ++j) {
      float v = acc[t][j];
      int m = mrow + j;
      int n = ncol + t * 16;
      if constexpr (MODE == 0) {
        ((u16*)Cout)[(long)blockIdx.z * sC + (long)m * ldc + n] = f32_to_bf16_rne(v);
      } else if constexpr (MODE == 1) {
        ((float*)Cout)[(long)blockIdx.z * sC + (long)m * ldc + n] = v;
      } else if constexpr (MODE == 2) {
        int b = m >> 11, s = m & (S_LEN - 1);
        ((u16*)Cout)[(long)b * (DIM * S_LEN) + (long)n * S_LEN + s] = f32_to_bf16_rne(v);
      } else if constexpr (MODE == 3) {
        int b = m >> 11, s = m & (S_LEN - 1);
        ((float*)Cout)[((long)s * BATCH + b) * DIM + n] = v + bias[n];
      } else {  // MODE 4
        float r = v + bias[n];
        r = r > 0.f ? r : 0.f;
        ((u16*)Cout)[(long)m * ldc + n] = f32_to_bf16_rne(r);
      }
    }
  }
}

// ---------------------------------------------------------------- softmax (in-place, f32 -> bf16 P)
__global__ __launch_bounds__(256) void softmax_kernel(float* __restrict__ logits) {
  const long row = blockIdx.x;                      // 0 .. BATCH*S_LEN-1
  float* lr = logits + row * S_LEN;
  u16*   pr = (u16*)logits + row * (2 * S_LEN);     // bf16 row in low half of f32 row
  const int tid = threadIdx.x;
  __shared__ float red[256];

  float v[8];
  float mx = -3.4e38f;
#pragma unroll
  for (int i = 0; i < 8; ++i) {
    v[i] = lr[tid + i * 256] * SOFTMAX_SCALE;
    mx = fmaxf(mx, v[i]);
  }
  red[tid] = mx; __syncthreads();
  for (int s = 128; s > 0; s >>= 1) { if (tid < s) red[tid] = fmaxf(red[tid], red[tid + s]); __syncthreads(); }
  mx = red[0]; __syncthreads();

  float sum = 0.f;
#pragma unroll
  for (int i = 0; i < 8; ++i) { v[i] = __expf(v[i] - mx); sum += v[i]; }
  red[tid] = sum; __syncthreads();
  for (int s = 128; s > 0; s >>= 1) { if (tid < s) red[tid] += red[tid + s]; __syncthreads(); }
  const float inv = 1.0f / red[0];
  __syncthreads();

#pragma unroll
  for (int i = 0; i < 8; ++i) pr[tid + i * 256] = f32_to_bf16_rne(v[i] * inv);
}

// ---------------------------------------------------------------- LayerNorm (f32 res -> bf16 h)
__global__ __launch_bounds__(256) void layernorm_kernel(const float* __restrict__ res,
                                                        u16* __restrict__ h,
                                                        const float* __restrict__ g,
                                                        const float* __restrict__ beta) {
  const long row = blockIdx.x;                      // 0 .. BATCH*S_LEN-1
  const float* r = res + row * DIM;
  const int tid = threadIdx.x;
  __shared__ float red[256];

  float x0 = r[tid], x1 = r[tid + 256];
  red[tid] = x0 + x1; __syncthreads();
  for (int s = 128; s > 0; s >>= 1) { if (tid < s) red[tid] += red[tid + s]; __syncthreads(); }
  const float mean = red[0] * (1.0f / DIM); __syncthreads();

  float d0 = x0 - mean, d1 = x1 - mean;
  red[tid] = d0 * d0 + d1 * d1; __syncthreads();
  for (int s = 128; s > 0; s >>= 1) { if (tid < s) red[tid] += red[tid + s]; __syncthreads(); }
  const float rstd = rsqrtf(red[0] * (1.0f / DIM) + LN_EPS);

  h[row * DIM + tid]       = f32_to_bf16_rne(d0 * rstd * g[tid]       + beta[tid]);
  h[row * DIM + tid + 256] = f32_to_bf16_rne(d1 * rstd * g[tid + 256] + beta[tid + 256]);
}

// ---------------------------------------------------------------- workspace layout (bytes)
static constexpr size_t SZ_ACT = (size_t)BATCH * S_LEN * DIM * 2;      // 8 MB bf16
static constexpr size_t SZ_W   = (size_t)DIM * DIM * 2;                // 512 KB bf16
static constexpr size_t SZ_LOG = (size_t)BATCH * S_LEN * S_LEN * 4;    // 64 MB f32

static constexpr size_t OFF_XBF = 0;
static constexpr size_t OFF_WQ  = OFF_XBF + SZ_ACT;
static constexpr size_t OFF_WK  = OFF_WQ + SZ_W;
static constexpr size_t OFF_WV  = OFF_WK + SZ_W;
static constexpr size_t OFF_W1  = OFF_WV + SZ_W;
static constexpr size_t OFF_W2  = OFF_W1 + SZ_W;
static constexpr size_t OFF_Q   = OFF_W2 + SZ_W;
static constexpr size_t OFF_K   = OFF_Q + SZ_ACT;
static constexpr size_t OFF_VT  = OFF_K + SZ_ACT;
static constexpr size_t OFF_LOG = OFF_VT + SZ_ACT;
static constexpr size_t OFF_RES = OFF_LOG + SZ_LOG;
// reuse: h -> OFF_Q, h2 -> OFF_K (q/k dead after logits GEMM)

extern "C" void kernel_launch(void* const* d_in, const int* in_sizes, int n_in,
                              void* d_out, int out_size, void* d_ws, size_t ws_size,
                              hipStream_t stream) {
  const float* x    = (const float*)d_in[0];
  const float* Wq   = (const float*)d_in[1];
  const float* Wk   = (const float*)d_in[2];
  const float* Wv   = (const float*)d_in[3];
  const float* ln_g = (const float*)d_in[4];
  const float* ln_b = (const float*)d_in[5];
  const float* W1   = (const float*)d_in[6];
  const float* b1   = (const float*)d_in[7];
  const float* W2   = (const float*)d_in[8];
  const float* b2   = (const float*)d_in[9];
  (void)in_sizes; (void)n_in; (void)out_size; (void)ws_size;

  char* ws = (char*)d_ws;
  u16*   xbf    = (u16*)(ws + OFF_XBF);
  u16*   wq     = (u16*)(ws + OFF_WQ);
  u16*   wk     = (u16*)(ws + OFF_WK);
  u16*   wv     = (u16*)(ws + OFF_WV);
  u16*   w1     = (u16*)(ws + OFF_W1);
  u16*   w2     = (u16*)(ws + OFF_W2);
  u16*   q      = (u16*)(ws + OFF_Q);
  u16*   k      = (u16*)(ws + OFF_K);
  u16*   vT     = (u16*)(ws + OFF_VT);
  float* logits = (float*)(ws + OFF_LOG);
  float* res    = (float*)(ws + OFF_RES);
  u16*   h      = (u16*)(ws + OFF_Q);   // reuse
  u16*   h2     = (u16*)(ws + OFF_K);   // reuse
  float* out    = (float*)d_out;

  const int NW = DIM * DIM;  // 262144

  // 1) fp32 -> bf16 conversions (x also transposed to batch-first)
  cvt_x_kernel<<<(BATCH * S_LEN * DIM) / 256, 256, 0, stream>>>(x, xbf);
  cvt_w_kernel<<<NW / 256, 256, 0, stream>>>(Wq, wq, NW);
  cvt_w_kernel<<<NW / 256, 256, 0, stream>>>(Wk, wk, NW);
  cvt_w_kernel<<<NW / 256, 256, 0, stream>>>(Wv, wv, NW);
  cvt_w_kernel<<<NW / 256, 256, 0, stream>>>(W1, w1, NW);
  cvt_w_kernel<<<NW / 256, 256, 0, stream>>>(W2, w2, NW);

  const int M_ALL = BATCH * S_LEN;  // 8192
  // 2) QKV projections: (8192x512) @ (512x512)^T
  gemm_bf16_kernel<0><<<dim3(DIM / 64, M_ALL / 128, 1), 256, 0, stream>>>(
      xbf, 0, wq, 0, q, 0, DIM, DIM, DIM, DIM, nullptr);
  gemm_bf16_kernel<0><<<dim3(DIM / 64, M_ALL / 128, 1), 256, 0, stream>>>(
      xbf, 0, wk, 0, k, 0, DIM, DIM, DIM, DIM, nullptr);
  gemm_bf16_kernel<2><<<dim3(DIM / 64, M_ALL / 128, 1), 256, 0, stream>>>(
      xbf, 0, wv, 0, vT, 0, DIM, DIM, DIM, 0, nullptr);

  // 3) logits = q @ k^T per batch: (2048x2048x512), f32 out
  gemm_bf16_kernel<1><<<dim3(S_LEN / 64, S_LEN / 128, BATCH), 256, 0, stream>>>(
      q, (long)S_LEN * DIM, k, (long)S_LEN * DIM,
      logits, (long)S_LEN * S_LEN, DIM, DIM, DIM, S_LEN, nullptr);

  // 4) softmax rows (scale applied here), bf16 P written in-place
  softmax_kernel<<<BATCH * S_LEN, 256, 0, stream>>>(logits);

  // 5) res = P @ V per batch: (2048x512x2048); P rows strided 2*S_LEN u16
  gemm_bf16_kernel<1><<<dim3(DIM / 64, S_LEN / 128, BATCH), 256, 0, stream>>>(
      (const u16*)logits, (long)S_LEN * (2 * S_LEN), vT, (long)DIM * S_LEN,
      res, (long)S_LEN * DIM, S_LEN, 2 * S_LEN, S_LEN, DIM, nullptr);

  // 6) LayerNorm -> bf16 h
  layernorm_kernel<<<BATCH * S_LEN, 256, 0, stream>>>(res, h, ln_g, ln_b);

  // 7) FFN1: relu(h @ W1^T + b1) -> bf16 h2
  gemm_bf16_kernel<4><<<dim3(DIM / 64, M_ALL / 128, 1), 256, 0, stream>>>(
      h, 0, w1, 0, h2, 0, DIM, DIM, DIM, DIM, b1);

  // 8) FFN2: h2 @ W2^T + b2 -> f32 out, transposed back to (S,B,D)
  gemm_bf16_kernel<3><<<dim3(DIM / 64, M_ALL / 128, 1), 256, 0, stream>>>(
      h2, 0, w2, 0, out, 0, DIM, DIM, DIM, 0, b2);
}
